// scAG_34419867910930
// MI455X (gfx1250) — matrix operand order repaired and checked by reading, not compiled, not run on hardware
//
#include <hip/hip_runtime.h>
#include <hip/hip_bf16.h>
#include <cmath>

// ---------------- problem constants ----------------
#define BB   4
#define CE_  128
#define CD_  256
#define HO   256
#define WO   256
#define HD_  128
#define WD_  128
#define HW   (HO * WO)       // 65536
#define HWD  (HD_ * WD_)     // 16384
#define NN   16

typedef __attribute__((ext_vector_type(16))) _Float16 v16h;
typedef __attribute__((ext_vector_type(8)))  float    v8f;

// ---------------- reduction helpers (wave32) ----------------
__device__ __forceinline__ float waveSum(float v) {
    #pragma unroll
    for (int o = 16; o > 0; o >>= 1) v += __shfl_xor(v, o, 32);
    return v;
}
__device__ __forceinline__ float waveMax(float v) {
    #pragma unroll
    for (int o = 16; o > 0; o >>= 1) v = fmaxf(v, __shfl_xor(v, o, 32));
    return v;
}

__device__ __forceinline__ void blockReduceWrite(float s, float m, float* av, float* mx,
                                                 int idx, float invCount) {
    __shared__ float ls[8], lm[8];
    int wid = threadIdx.x >> 5, lid = threadIdx.x & 31;
    s = waveSum(s);
    m = waveMax(m);
    if (lid == 0) { ls[wid] = s; lm[wid] = m; }
    __syncthreads();
    if (threadIdx.x == 0) {
        float S = 0.f, M = -3.4e38f;
        #pragma unroll
        for (int i = 0; i < 8; i++) { S += ls[i]; M = fmaxf(M, lm[i]); }
        av[idx] = S * invCount;
        mx[idx] = M;
    }
}

// ---------------- CAG enc: per-channel mean/max over 256x256 ----------------
// grid = BB*CE_ (one channel per block), fully coalesced float4 streaming.
__global__ __launch_bounds__(256) void k_cag_enc(const float* __restrict__ xe,
                                                 float* __restrict__ av,
                                                 float* __restrict__ mx) {
    int bc = blockIdx.x;                       // b*CE_ + c
    const float4* p4 = (const float4*)(xe + (size_t)bc * HW);
    float s = 0.f, m = -3.4e38f;
    for (int i = threadIdx.x; i < HW / 4; i += 256) {
        float4 v = p4[i];
        s += v.x + v.y + v.z + v.w;
        m = fmaxf(m, fmaxf(fmaxf(v.x, v.y), fmaxf(v.z, v.w)));
    }
    blockReduceWrite(s, m, av, mx, bc, 1.f / (float)HW);
}

// ---------------- CAG dec: per-channel mean/max of relu(up2(x_dec)) ----------
// grid = BB*CD_. The 64KB input channel is staged into LDS with the CDNA5
// async Global->LDS copy engine (ASYNCcnt, no VGPR bounce), then the 256x256
// upsampled field is evaluated on the fly (never materialized).
__global__ __launch_bounds__(256) void k_cag_dec(const float* __restrict__ xd,
                                                 float* __restrict__ av,
                                                 float* __restrict__ mx) {
    __shared__ __align__(16) float tile[HWD];  // 64 KB (320 KB/WGP available)
    int bc = blockIdx.x;                       // b*CD_ + c
    const float* src = xd + (size_t)bc * HWD;
    // 16B per lane per issue; generic pointers to LDS carry the LDS byte
    // offset in their low 32 bits (aperture tag lives in addr[63:32]).
    for (int i = threadIdx.x * 4; i < HWD; i += 256 * 4) {
        unsigned lds_off = (unsigned)(uintptr_t)&tile[i];
        unsigned long long gaddr = (unsigned long long)(uintptr_t)(src + i);
        asm volatile("global_load_async_to_lds_b128 %0, %1, off"
                     :: "v"(lds_off), "v"(gaddr) : "memory");
    }
    asm volatile("s_wait_asynccnt 0" ::: "memory");
    __syncthreads();                           // all waves' copies visible

    const float SC = 127.0f / 255.0f;          // align_corners 2x scale
    int x = threadIdx.x;
    float px = x * SC;
    int ix0 = min((int)px, WD_ - 1);
    int ix1 = min(ix0 + 1, WD_ - 1);
    float wx = px - (float)ix0;

    float s = 0.f, m = -3.4e38f;
    for (int y = 0; y < HO; y++) {
        float py = y * SC;
        int iy0 = min((int)py, HD_ - 1);
        int iy1 = min(iy0 + 1, HD_ - 1);
        float wy = py - (float)iy0;
        float v00 = tile[iy0 * WD_ + ix0], v01 = tile[iy0 * WD_ + ix1];
        float v10 = tile[iy1 * WD_ + ix0], v11 = tile[iy1 * WD_ + ix1];
        float t0 = v00 + (v01 - v00) * wx;
        float t1 = v10 + (v11 - v10) * wx;
        float v  = fmaxf(t0 + (t1 - t0) * wy, 0.f);   // relu
        s += v;
        m = fmaxf(m, v);
    }
    blockReduceWrite(s, m, av, mx, bc, 1.f / (float)HW);
}

// ---------------- SAG enc: per-pixel channel max/mean/dot -> 3-plane map -----
// grid = BB*HO (one image row per block, thread = x). Coalesced per channel.
__global__ __launch_bounds__(256) void k_sag_enc(const float* __restrict__ xe,
                                                 const float* __restrict__ sw1,
                                                 const float* __restrict__ sb1,
                                                 float* __restrict__ se) {
    int row = blockIdx.x;
    int b = row >> 8, y = row & 255, x = threadIdx.x;
    size_t pix = (size_t)y * WO + x;
    const float* base = xe + (size_t)b * CE_ * HW + pix;
    float m = -3.4e38f, s = 0.f, d = 0.f;
    #pragma unroll 4
    for (int c = 0; c < CE_; c++) {
        float v = base[(size_t)c * HW];
        m = fmaxf(m, v);
        s += v;
        d = fmaf(v, sw1[c], d);                 // sw1[c] uniform -> s_load
    }
    size_t sb = (size_t)(b * 3) * HW + pix;
    se[sb]          = m;
    se[sb + HW]     = s * (1.f / (float)CE_);
    se[sb + 2 * HW] = d + sb1[0];
}

// ---------------- SAG dec: same, on-the-fly relu(up2(x_dec)) -----------------
__global__ __launch_bounds__(256) void k_sag_dec(const float* __restrict__ xd,
                                                 const float* __restrict__ sw1,
                                                 const float* __restrict__ sb1,
                                                 float* __restrict__ sd) {
    int row = blockIdx.x;
    int b = row >> 8, y = row & 255, x = threadIdx.x;
    const float SC = 127.0f / 255.0f;
    float py = y * SC;
    int iy0 = min((int)py, HD_ - 1);
    int iy1 = min(iy0 + 1, HD_ - 1);
    float wy = py - (float)iy0;
    float px = x * SC;
    int ix0 = min((int)px, WD_ - 1);
    int ix1 = min(ix0 + 1, WD_ - 1);
    float wx = px - (float)ix0;

    int o00 = iy0 * WD_ + ix0, o01 = iy0 * WD_ + ix1;
    int o10 = iy1 * WD_ + ix0, o11 = iy1 * WD_ + ix1;
    const float* base = xd + (size_t)b * CD_ * HWD;

    float m = -3.4e38f, s = 0.f, d = 0.f;
    #pragma unroll 2
    for (int c = 0; c < CD_; c++) {
        const float* p = base + (size_t)c * HWD;
        float v00 = p[o00], v01 = p[o01], v10 = p[o10], v11 = p[o11];
        float t0 = v00 + (v01 - v00) * wx;
        float t1 = v10 + (v11 - v10) * wx;
        float v  = fmaxf(t0 + (t1 - t0) * wy, 0.f);
        m = fmaxf(m, v);
        s += v;
        d = fmaf(v, sw1[c], d);
    }
    size_t pix = (size_t)y * WO + x;
    size_t sb = (size_t)(b * 3) * HW + pix;
    sd[sb]          = m;
    sd[sb + HW]     = s * (1.f / (float)CD_);
    sd[sb + 2 * HW] = d + sb1[0];
}

// ---------------- CAG combine: tiny GEMM chain on WMMA ----------------------
// Single wave32. ce+cd = (16x16 padded)  [av|mx](4xC) @ w.T(CxN)  via
// v_wmma_f32_16x16x32_f16 K-tiling, then  g(4x16) @ cw_f.T(16x128)  (K padded
// to 32) with LDS re-staging from D-layout to A-layout.
// Row padding is done branchlessly: clamp the row index (load always safe),
// then select the value -> v_cndmask, no EXEC gymnastics around loads.
__device__ __forceinline__ int a_kidx(int e, int hf) {
    int r = e >> 1, p = e & 1;
    return (r < 4) ? (8 * hf + 2 * r + p) : (16 + 8 * hf + 2 * (r - 4) + p);
}

__global__ __launch_bounds__(32) void k_combine(
    const float* __restrict__ eav, const float* __restrict__ emx,
    const float* __restrict__ dav, const float* __restrict__ dmx,
    const float* __restrict__ cw_ea, const float* __restrict__ cb_ea,
    const float* __restrict__ cw_em, const float* __restrict__ cb_em,
    const float* __restrict__ cw_da, const float* __restrict__ cb_da,
    const float* __restrict__ cw_dm, const float* __restrict__ cb_dm,
    const float* __restrict__ cw_f,  const float* __restrict__ cb_f,
    float* __restrict__ ch) {
    const int lane = threadIdx.x;
    const int mr = lane & 15;            // M row (batch, rows >= BB are padding)
    const int hf = lane >> 4;
    const int n  = lane & 15;            // N column
    const float msk = (mr < BB) ? 1.f : 0.f;
    const int rm = (mr < BB) ? mr : 0;   // clamped -> loads always in-bounds

    v8f acc = {0.f, 0.f, 0.f, 0.f, 0.f, 0.f, 0.f, 0.f};

    // encoder branch: av @ cw_ea.T + mx @ cw_em.T   (K = 128)
    #pragma unroll
    for (int kb = 0; kb < CE_; kb += 32) {
        v16h a_av, a_mx, b_wa, b_wm;
        #pragma unroll
        for (int e = 0; e < 16; e++) {
            int c  = kb + a_kidx(e, hf);
            a_av[e] = (_Float16)(eav[rm * CE_ + c] * msk);
            a_mx[e] = (_Float16)(emx[rm * CE_ + c] * msk);
            int kB = kb + a_kidx(e, hf);
            b_wa[e] = (_Float16)cw_ea[n * CE_ + kB];
            b_wm[e] = (_Float16)cw_em[n * CE_ + kB];
        }
        acc = __builtin_amdgcn_wmma_f32_16x16x32_f16(false, a_av, false, b_wa,
                                                     (short)0, acc, false, false);
        acc = __builtin_amdgcn_wmma_f32_16x16x32_f16(false, a_mx, false, b_wm,
                                                     (short)0, acc, false, false);
    }
    // decoder branch: av @ cw_da.T + mx @ cw_dm.T   (K = 256)
    #pragma unroll
    for (int kb = 0; kb < CD_; kb += 32) {
        v16h a_av, a_mx, b_wa, b_wm;
        #pragma unroll
        for (int e = 0; e < 16; e++) {
            int c  = kb + a_kidx(e, hf);
            a_av[e] = (_Float16)(dav[rm * CD_ + c] * msk);
            a_mx[e] = (_Float16)(dmx[rm * CD_ + c] * msk);
            int kB = kb + a_kidx(e, hf);
            b_wa[e] = (_Float16)cw_da[n * CD_ + kB];
            b_wm[e] = (_Float16)cw_dm[n * CD_ + kB];
        }
        acc = __builtin_amdgcn_wmma_f32_16x16x32_f16(false, a_av, false, b_wa,
                                                     (short)0, acc, false, false);
        acc = __builtin_amdgcn_wmma_f32_16x16x32_f16(false, a_mx, false, b_wm,
                                                     (short)0, acc, false, false);
    }

    // g = ce + cd + biases ; re-stage D-layout -> A-layout through LDS
    float bias = cb_ea[n] + cb_em[n] + cb_da[n] + cb_dm[n];
    __shared__ float g[16][16];
    #pragma unroll
    for (int r = 0; r < 8; r++) g[r + 8 * hf][n] = acc[r] + bias;
    __syncthreads();

    // ch = sigmoid(g @ cw_f.T + cb_f) : K=16 zero-padded to 32, 8 N-tiles of 16
    #pragma unroll
    for (int t = 0; t < 8; t++) {
        v16h a2, b2;
        #pragma unroll
        for (int e = 0; e < 16; e++) {
            int k = a_kidx(e, hf);
            float va = g[mr][k & (NN - 1)];                  // safe LDS read
            a2[e] = (_Float16)((k < NN) ? va : 0.f);
            int kB = a_kidx(e, hf);
            float vb = cw_f[(t * 16 + n) * NN + (kB & (NN - 1))];
            b2[e] = (_Float16)((kB < NN) ? vb : 0.f);
        }
        v8f acc2 = {0.f, 0.f, 0.f, 0.f, 0.f, 0.f, 0.f, 0.f};
        acc2 = __builtin_amdgcn_wmma_f32_16x16x32_f16(false, a2, false, b2,
                                                      (short)0, acc2, false, false);
        if (hf == 0) {                          // rows 0..7 live in lanes 0..15
            float bf = cb_f[t * 16 + n];
            #pragma unroll
            for (int r = 0; r < BB; r++) {
                float v = acc2[r] + bf;
                ch[r * CE_ + t * 16 + n] = 1.f / (1.f + __expf(-v));
            }
        }
    }
}

// ---------------- final: conv7(se)+conv7(sd) -> sigmoid -> broadcast mul ----
// grid = BB*HO (row per block). spat computed once per pixel, reused over the
// 128-channel stream; x_enc read is L2-warm, out write is the only cold store.
__global__ __launch_bounds__(256) void k_final(const float* __restrict__ xe,
                                               const float* __restrict__ se,
                                               const float* __restrict__ sd,
                                               const float* __restrict__ sw2e,
                                               const float* __restrict__ sb2e,
                                               const float* __restrict__ sw2d,
                                               const float* __restrict__ sb2d,
                                               const float* __restrict__ ch,
                                               float* __restrict__ out) {
    int row = blockIdx.x;
    int b = row >> 8, y = row & 255, x = threadIdx.x;

    float acc = sb2e[0] + sb2d[0];
    const float* seb = se + (size_t)(b * 3) * HW;
    const float* sdb = sd + (size_t)(b * 3) * HW;
    #pragma unroll
    for (int ci = 0; ci < 3; ci++) {
        #pragma unroll
        for (int ky = 0; ky < 7; ky++) {
            int yy = y + ky - 3;
            if ((unsigned)yy >= HO) continue;   // zero padding
            #pragma unroll
            for (int kx = 0; kx < 7; kx++) {
                int xx = x + kx - 3;
                if ((unsigned)xx >= WO) continue;
                size_t off = (size_t)ci * HW + (size_t)yy * WO + xx;
                float we = sw2e[ci * 49 + ky * 7 + kx];   // uniform -> s_load
                float wd = sw2d[ci * 49 + ky * 7 + kx];
                acc = fmaf(seb[off], we, acc);
                acc = fmaf(sdb[off], wd, acc);
            }
        }
    }
    float spat = 1.f / (1.f + __expf(-acc));

    size_t pix = (size_t)y * WO + x;
    const float* xb  = xe  + (size_t)b * CE_ * HW + pix;
    float*       ob  = out + (size_t)b * CE_ * HW + pix;
    const float* chb = ch + b * CE_;
    #pragma unroll 4
    for (int c = 0; c < CE_; c++) {
        if (c + 1 < CE_) __builtin_prefetch(xb + (size_t)(c + 1) * HW, 0, 0);
        ob[(size_t)c * HW] = xb[(size_t)c * HW] * spat * chb[c];
    }
}

// ---------------- host launcher ----------------
extern "C" void kernel_launch(void* const* d_in, const int* in_sizes, int n_in,
                              void* d_out, int out_size, void* d_ws, size_t ws_size,
                              hipStream_t stream) {
    const float* x_enc = (const float*)d_in[0];
    const float* x_dec = (const float*)d_in[1];
    const float* cw_ea = (const float*)d_in[2];
    const float* cb_ea = (const float*)d_in[3];
    const float* cw_em = (const float*)d_in[4];
    const float* cb_em = (const float*)d_in[5];
    const float* cw_da = (const float*)d_in[6];
    const float* cb_da = (const float*)d_in[7];
    const float* cw_dm = (const float*)d_in[8];
    const float* cb_dm = (const float*)d_in[9];
    const float* cw_f  = (const float*)d_in[10];
    const float* cb_f  = (const float*)d_in[11];
    const float* sw1_e = (const float*)d_in[12];
    const float* sb1_e = (const float*)d_in[13];
    const float* sw2_e = (const float*)d_in[14];
    const float* sb2_e = (const float*)d_in[15];
    const float* sw1_d = (const float*)d_in[16];
    const float* sb1_d = (const float*)d_in[17];
    const float* sw2_d = (const float*)d_in[18];
    const float* sb2_d = (const float*)d_in[19];
    float* out = (float*)d_out;

    // workspace layout (floats): all buffers fully written before read
    float* ws  = (float*)d_ws;
    float* eav = ws;                       // BB*CE_          = 512
    float* emx = ws + 512;                 // BB*CE_          = 512
    float* dav = ws + 1024;                // BB*CD_          = 1024
    float* dmx = ws + 2048;                // BB*CD_          = 1024
    float* chg = ws + 3072;                // BB*CE_          = 512
    float* se  = ws + 4096;                // BB*3*HW         = 786432
    float* sd  = se + (size_t)BB * 3 * HW; // BB*3*HW         = 786432

    // Order chosen for L2 residency: x_enc passes back-to-back, then x_dec.
    k_cag_enc<<<BB * CE_, 256, 0, stream>>>(x_enc, eav, emx);
    k_sag_enc<<<BB * HO,  256, 0, stream>>>(x_enc, sw1_e, sb1_e, se);
    k_cag_dec<<<BB * CD_, 256, 0, stream>>>(x_dec, dav, dmx);
    k_sag_dec<<<BB * HO,  256, 0, stream>>>(x_dec, sw1_d, sb1_d, sd);
    k_combine<<<1, 32, 0, stream>>>(eav, emx, dav, dmx,
                                    cw_ea, cb_ea, cw_em, cb_em,
                                    cw_da, cb_da, cw_dm, cb_dm,
                                    cw_f, cb_f, chg);
    k_final<<<BB * HO, 256, 0, stream>>>(x_enc, se, sd,
                                         sw2_e, sb2_e, sw2_d, sb2_d, chg, out);
}
